// GCNNet_56049323213279
// MI455X (gfx1250) — matrix operand order repaired and checked by reading, not compiled, not run on hardware
//
#include <hip/hip_runtime.h>

typedef __attribute__((ext_vector_type(16))) __bf16 v16bf;
typedef __attribute__((ext_vector_type(8)))  float  v8f;

#define N_NODES 10000

// fp32 -> bf16 round-to-nearest-even (bit-level)
static __device__ __forceinline__ __bf16 f2bf(float f) {
    unsigned u = __builtin_bit_cast(unsigned, f);
    unsigned bias = 0x7fffu + ((u >> 16) & 1u);
    unsigned short h = (unsigned short)((u + bias) >> 16);
    return __builtin_bit_cast(__bf16, h);
}

// ---------------------------------------------------------------------------
// Degree / normalization
// ---------------------------------------------------------------------------
__global__ void deg_init(float* deg, int n) {
    int i = blockIdx.x * blockDim.x + threadIdx.x;
    if (i < n) deg[i] = 1.0f;                 // self-loop
}

__global__ void deg_accum(const int* __restrict__ dst, float* deg, int e) {
    int i = blockIdx.x * blockDim.x + threadIdx.x;
    if (i < e) atomicAdd(&deg[dst[i]], 1.0f);
}

__global__ void deg_finish(float* deg, int n) {
    int i = blockIdx.x * blockDim.x + threadIdx.x;
    if (i < n) deg[i] = rsqrtf(deg[i]);       // deg >= 1 always
}

// ---------------------------------------------------------------------------
// Pack W [K, NCOLS] fp32 -> bf16, chunked + transposed to match the GEMM's
// LDS tile layout: element (chunk c, col n, k) at c*NCOLS*32 + n*32 + k.
// Tail chunk zero-padded, so the GEMM B-side needs no bounds checks at all.
// ---------------------------------------------------------------------------
template <int NCOLS>
__global__ void pack_w(const float* __restrict__ W, __bf16* __restrict__ out,
                       int K, int nchunks) {
    int i = blockIdx.x * blockDim.x + threadIdx.x;
    if (i >= nchunks * NCOLS * 32) return;
    int c = i / (NCOLS * 32);
    int r = i % (NCOLS * 32);
    int n = r / 32, k = r % 32;
    int kg = c * 32 + k;
    float f = (kg < K) ? W[(long)kg * NCOLS + n] : 0.0f;
    out[i] = f2bf(f);
}

// ---------------------------------------------------------------------------
// WMMA GEMM:  C[M, CT*16] = op(A[M, K]) @ W[K, CT*16], op = optional ReLU.
// 8 waves/block, one 16-row tile per wave (128 rows/block).
// B: pre-packed bf16 buffer -> double-buffered global_load_async_to_lds_b128
//    (ASYNCcnt), overlapping WMMA work; all CT fragments loaded into registers
//    before the WMMA burst so the DS loads clause behind one s_wait_dscnt.
// A: guard-free global_load_b128 stream (guards only in the single tail chunk).
// ---------------------------------------------------------------------------
template <int CT, bool RELU>
__global__ __launch_bounds__(256) void wmma_gemm(
        const float* __restrict__ A, const __bf16* __restrict__ Wp,
        float* __restrict__ C, int M, int K) {
    constexpr int NCOLS       = CT * 16;
    constexpr int CHUNK_ELEMS = NCOLS * 32;        // bf16 elems per K-chunk
    constexpr int CHUNK_BYTES = CHUNK_ELEMS * 2;   // 1/2/4 KB
    constexpr int LOADERS     = CHUNK_BYTES / 16;  // one b128 per loader lane

    __shared__ __align__(64) __bf16 ldsW[2][CHUNK_ELEMS];

    const int tid     = threadIdx.x;
    const int lane    = tid & 31;
    const int wave    = tid >> 5;
    const int rowTile = blockIdx.x * 8 + wave;

    const int  arow  = rowTile * 16 + (lane & 15);          // A row this lane
    const long arowC = (long)((arow < M) ? arow : 0) * K;   // clamped base
    const int  khiA  = (lane >> 4) * 16;                    // A half-wave K off
    const int  khiB  = (lane >> 4) * 16;                    // B half-wave K off
    const int  ncol  = lane & 15;

    const int nchunks = (K + 31) >> 5;

    // raw LDS byte offsets (flat-LDS aperture: low 32 bits of generic ptr)
    const unsigned ldsBase[2] = {
        (unsigned)(unsigned long long)&ldsW[0][0],
        (unsigned)(unsigned long long)&ldsW[1][0]
    };

    auto issue_async = [&](int c, int buf) {
        if (tid < LOADERS) {
            unsigned dst  = ldsBase[buf] + tid * 16;
            unsigned voff = (unsigned)c * CHUNK_BYTES + tid * 16;
            asm volatile("global_load_async_to_lds_b128 %0, %1, %2"
                         :: "v"(dst), "v"(voff), "s"(Wp)
                         : "memory");
        }
    };

    v8f acc[CT] = {};

    issue_async(0, 0);

    for (int c = 0; c < nchunks; ++c) {
        const int cur = c & 1;
        const int k0  = c * 32;

        asm volatile("s_wait_asynccnt 0x0" ::: "memory");  // my chunk-c copy done
        __syncthreads();                                   // all copies done; c-1 consumed

        if (c + 1 < nchunks) issue_async(c + 1, cur ^ 1);  // overlap with compute

        // ---- A fragment ----
        v16bf a;
        if (k0 + 32 <= K) {    // uniform branch; fast path has zero guards
            const float4* p0 = (const float4*)(A + arowC + k0 + khiA);
            const float4* p1 = (const float4*)(A + arowC + k0 + khiA + 8);
            float4 q[4] = { p0[0], p0[1], p1[0], p1[1] };
#pragma unroll
            for (int j = 0; j < 4; ++j) {
                float f0 = q[j].x, f1 = q[j].y, f2 = q[j].z, f3 = q[j].w;
                if (RELU) {
                    f0 = fmaxf(f0, 0.f); f1 = fmaxf(f1, 0.f);
                    f2 = fmaxf(f2, 0.f); f3 = fmaxf(f3, 0.f);
                }
                a[4 * j + 0] = f2bf(f0); a[4 * j + 1] = f2bf(f1);
                a[4 * j + 2] = f2bf(f2); a[4 * j + 3] = f2bf(f3);
            }
        } else {               // tail chunk (at most once per kernel)
#pragma unroll
            for (int i = 0; i < 16; ++i) {
                int k = k0 + khiA + i;
                float f = (k < K) ? A[arowC + k] : 0.0f;
                if (RELU) f = fmaxf(f, 0.0f);
                a[i] = f2bf(f);
            }
        }

        // ---- load ALL B fragments first (DS loads clause together), then
        //      issue the WMMA burst back-to-back on independent accumulators --
        const __bf16* bbase = &ldsW[cur][0];
        v16bf bfrag[CT];
#pragma unroll
        for (int t = 0; t < CT; ++t)
            bfrag[t] = *(const v16bf*)(bbase + (t * 16 + ncol) * 32 + khiB);
#pragma unroll
        for (int t = 0; t < CT; ++t)
            acc[t] = __builtin_amdgcn_wmma_f32_16x16x32_bf16(
                false, a, false, bfrag[t], (short)0, acc[t], false, false);
    }

    // ---- store (32-bit C/D layout: VGPR r -> M=r (lanes 0-15) / M=8+r) ----
    const int srow0 = rowTile * 16 + (lane >> 4) * 8;
#pragma unroll
    for (int t = 0; t < CT; ++t) {
#pragma unroll
        for (int r = 0; r < 8; ++r) {
            int rr = srow0 + r;
            if (rr < M) C[(long)rr * NCOLS + t * 16 + ncol] = acc[t][r];
        }
    }
}

// ---------------------------------------------------------------------------
// GCN aggregation: deterministic init (bias + self-loop), then atomic scatter
// ---------------------------------------------------------------------------
template <int F>
__global__ void conv_init(const float* __restrict__ h,
                          const float* __restrict__ dinv,
                          const float* __restrict__ bias,
                          float* __restrict__ out, int n) {
    int i = blockIdx.x * blockDim.x + threadIdx.x;
    if (i >= n * F) return;
    int node = i / F, f = i % F;
    float d = dinv[node];
    out[i] = bias[f] + h[i] * d * d;
}

template <int F>
__global__ void conv_edges(const float* __restrict__ h,
                           const float* __restrict__ dinv,
                           const int* __restrict__ src,
                           const int* __restrict__ dst,
                           float* __restrict__ out, int e) {
    constexpr int GROUPS = F / 4;
    int t  = blockIdx.x * blockDim.x + threadIdx.x;
    int ed = t / GROUPS;
    int fg = t % GROUPS;
    if (ed >= e) return;
    int s = src[ed], d = dst[ed];
    float norm = dinv[s] * dinv[d];
    const float4 v = *(const float4*)(h + (long)s * F + fg * 4);
    float* o = out + (long)d * F + fg * 4;
    atomicAdd(o + 0, v.x * norm);
    atomicAdd(o + 1, v.y * norm);
    atomicAdd(o + 2, v.z * norm);
    atomicAdd(o + 3, v.w * norm);
}

// ---------------------------------------------------------------------------
// Row softmax over 16 columns
// ---------------------------------------------------------------------------
__global__ void softmax16(const float* __restrict__ in, float* __restrict__ out,
                          int n) {
    int i = blockIdx.x * blockDim.x + threadIdx.x;
    if (i >= n) return;
    const float* r = in + (long)i * 16;
    float mx = r[0];
#pragma unroll
    for (int f = 1; f < 16; ++f) mx = fmaxf(mx, r[f]);
    float e[16], s = 0.0f;
#pragma unroll
    for (int f = 0; f < 16; ++f) { e[f] = __expf(r[f] - mx); s += e[f]; }
    float inv = 1.0f / s;
    float* o = out + (long)i * 16;
#pragma unroll
    for (int f = 0; f < 16; ++f) o[f] = e[f] * inv;
}

// ---------------------------------------------------------------------------
extern "C" void kernel_launch(void* const* d_in, const int* in_sizes, int n_in,
                              void* d_out, int out_size, void* d_ws, size_t ws_size,
                              hipStream_t stream) {
    const float* x   = (const float*)d_in[0];
    const int*   ei  = (const int*)d_in[1];   // [2, E] flat
    const float* W1  = (const float*)d_in[2];
    const float* b1  = (const float*)d_in[3];
    const float* W2  = (const float*)d_in[4];
    const float* b2  = (const float*)d_in[5];
    const float* W3  = (const float*)d_in[6];
    const float* b3  = (const float*)d_in[7];

    const int N = N_NODES;
    const int E = in_sizes[1] / 2;
    const int* srcP = ei;
    const int* dstP = ei + E;

    const int nc1 = (N + 31) / 32;     // 313 chunks, K = 10000
    const int nc2 = (64 + 31) / 32;    // 2
    const int nc3 = (32 + 31) / 32;    // 1

    // Workspace layout: [dinv][bufA][bufB][wp1][wp2][wp3]
    float*  dinv = (float*)d_ws;
    float*  bufA = dinv + 16384;
    float*  bufB = bufA + (size_t)N * 64;
    __bf16* wp1  = (__bf16*)(bufB + (size_t)N * 64);
    __bf16* wp2  = wp1 + (size_t)nc1 * 64 * 32;
    __bf16* wp3  = wp2 + (size_t)nc2 * 32 * 32;

    const int TB = 256;
    const int gemmBlocks = ((N + 15) / 16 + 7) / 8;   // 625 row tiles / 8 waves

    deg_init  <<<(N + TB - 1) / TB, TB, 0, stream>>>(dinv, N);
    deg_accum <<<(E + TB - 1) / TB, TB, 0, stream>>>(dstP, dinv, E);
    deg_finish<<<(N + TB - 1) / TB, TB, 0, stream>>>(dinv, N);

    // one-time bf16 packing of the three weight matrices
    pack_w<64><<<(nc1 * 64 * 32 + TB - 1) / TB, TB, 0, stream>>>(W1, wp1, N, nc1);
    pack_w<32><<<(nc2 * 32 * 32 + TB - 1) / TB, TB, 0, stream>>>(W2, wp2, 64, nc2);
    pack_w<16><<<(nc3 * 16 * 32 + TB - 1) / TB, TB, 0, stream>>>(W3, wp3, 32, nc3);

    // layer 1: h1 = x @ W1 (bandwidth-bound streaming GEMM), aggregate
    wmma_gemm<4, false><<<gemmBlocks, TB, 0, stream>>>(x, wp1, bufA, N, N);
    conv_init<64><<<(N * 64 + TB - 1) / TB, TB, 0, stream>>>(bufA, dinv, b1, bufB, N);
    conv_edges<64><<<((long)E * 16 + TB - 1) / TB, TB, 0, stream>>>(bufA, dinv, srcP, dstP, bufB, E);

    // layer 2: h2 = relu(agg1) @ W2, aggregate
    wmma_gemm<2, true><<<gemmBlocks, TB, 0, stream>>>(bufB, wp2, bufA, N, 64);
    conv_init<32><<<(N * 32 + TB - 1) / TB, TB, 0, stream>>>(bufA, dinv, b2, bufB, N);
    conv_edges<32><<<((long)E * 8 + TB - 1) / TB, TB, 0, stream>>>(bufA, dinv, srcP, dstP, bufB, E);

    // layer 3: h3 = relu(agg2) @ W3, aggregate, softmax
    wmma_gemm<1, true><<<gemmBlocks, TB, 0, stream>>>(bufB, wp3, bufA, N, 32);
    conv_init<16><<<(N * 16 + TB - 1) / TB, TB, 0, stream>>>(bufA, dinv, b3, bufB, N);
    conv_edges<16><<<((long)E * 4 + TB - 1) / TB, TB, 0, stream>>>(bufA, dinv, srcP, dstP, bufB, E);

    softmax16<<<(N + TB - 1) / TB, TB, 0, stream>>>(bufB, (float*)d_out, N);
}
